// DynamicLayer_70454643523808
// MI455X (gfx1250) — matrix operand order repaired and checked by reading, not compiled, not run on hardware
//
#include <hip/hip_runtime.h>
#include <hip/hip_bf16.h>

#define DEV __device__ __forceinline__

typedef __attribute__((ext_vector_type(16))) __bf16         bf16x16;
typedef __attribute__((ext_vector_type(8)))  float          f32x8;
typedef __attribute__((ext_vector_type(8)))  unsigned short u16x8;
typedef __attribute__((ext_vector_type(4)))  unsigned int   u32x4;
typedef __attribute__((ext_vector_type(8)))  unsigned int   u32x8;

union BF16Frag { bf16x16 v; u16x8 u[2]; };

// Problem constants (from reference)
constexpr int Bc  = 2;
constexpr int Sc  = 2048;
constexpr int Dc  = 1024;
constexpr int Hc  = 16;
constexpr int DHc = 64;
constexpr int Fc  = 4096;
constexpr int Kc  = 2048;     // selected tokens
constexpr int HDc = Hc * DHc; // 1024

DEV unsigned short f2bf(float f) {
  unsigned int u = __builtin_bit_cast(unsigned int, f);
  u += 0x7FFFu + ((u >> 16) & 1u);   // round-to-nearest-even
  return (unsigned short)(u >> 16);
}
DEV float bf2f(unsigned short s) {
  unsigned int u = ((unsigned int)s) << 16;
  return __builtin_bit_cast(float, u);
}

// A fragment: 16x32 bf16, row-major source; lane m=lane&15 half=lane>>4.
// elements j=0..15 <- K = 16*half + j
DEV bf16x16 load_a_frag(const unsigned short* rowbase, int half) {
  BF16Frag a;
  a.u[0] = *(const u16x8*)(rowbase + 16 * half);
  a.u[1] = *(const u16x8*)(rowbase + 16 * half + 8);
  return a.v;
}
// B fragment: 32x16 bf16; source gives 32 contiguous K elements for column n.
// elements j=0..15 <- K = 16*half + j
DEV bf16x16 load_b_frag(const unsigned short* colbase, int half) {
  BF16Frag b;
  const unsigned short* p = colbase + 16 * half;
  b.u[0] = *(const u16x8*)(p);
  b.u[1] = *(const u16x8*)(p + 8);
  return b.v;
}
DEV f32x8 wmma_bf16(bf16x16 a, bf16x16 b, f32x8 c) {
  return __builtin_amdgcn_wmma_f32_16x16x32_bf16(false, a, false, b, (short)0, c,
                                                 false, false);
}

// ---------------------------------------------------------------------------
// TDM: DMA a 64x64 bf16 tile (64 rows of Bt, 64 contiguous k-elements each)
// into LDS. D# packed per CDNA5 ISA 8.3/8.4 (count=1, type=2 "image",
// data_size=2B, tile_dim0=64, tile_dim1=64, tensor_dim0_stride=Kd).
// Issued by one wave; tracked with TENSORcnt.
// ---------------------------------------------------------------------------
DEV void tdm_load_b_tile(const unsigned short* gptr, unsigned lds_off, int Kd,
                         int N) {
  unsigned long long ga = (unsigned long long)(const void*)gptr;
  unsigned long long s0 = (unsigned long long)(unsigned)Kd;  // dim0 stride (elems)
  u32x4 g0 = {};
  g0[0] = 1u;                                        // count=1, user mode
  g0[1] = lds_off;                                   // lds_addr (bytes)
  g0[2] = (unsigned)(ga & 0xFFFFFFFFull);            // global_addr[31:0]
  g0[3] = (unsigned)((ga >> 32) & 0x1FFFFFFull)      // global_addr[56:32]
          | 0x80000000u;                             // type=2 (bits 127:126)
  u32x8 g1 = {};
  g1[0] = (1u << 16);                                // data_size=1 -> 2 bytes
  g1[1] = ((unsigned)Kd & 0xFFFFu) << 16;            // tensor_dim0[15:0]
  g1[2] = (((unsigned)Kd >> 16) & 0xFFFFu)           // tensor_dim0[31:16]
          | (((unsigned)N & 0xFFFFu) << 16);         // tensor_dim1[15:0]
  g1[3] = (((unsigned)N >> 16) & 0xFFFFu)            // tensor_dim1[31:16]
          | (64u << 16);                             // tile_dim0=64
  g1[4] = 64u;                                       // tile_dim1=64, tile_dim2=0
  g1[5] = (unsigned)(s0 & 0xFFFFFFFFull);            // tensor_dim0_stride[31:0]
  g1[6] = (unsigned)((s0 >> 32) & 0xFFFFull);        // stride[47:32], dim1_stride=0
  g1[7] = 0u;
  u32x4 gz = {};                                     // groups 2/3 unused (2D)
  asm volatile("tensor_load_to_lds %0, %1, %2, %3"
               :
               : "s"(g0), "s"(g1), "s"(gz), "s"(gz)
               : "memory");
}

// ---------------------------------------------------------------------------
// Weight convert + transpose: W[rows][cols] f32 -> Wt[cols][rows] bf16
// ---------------------------------------------------------------------------
__global__ __launch_bounds__(256) void convert_wt_kernel(
    const float* __restrict__ W, unsigned short* __restrict__ Wt,
    int rows, int cols) {
  size_t idx = (size_t)blockIdx.x * blockDim.x + threadIdx.x;
  if (idx >= (size_t)rows * cols) return;
  int r = (int)(idx / cols);
  int c = (int)(idx - (size_t)r * cols);
  Wt[(size_t)c * rows + r] = f2bf(W[idx]);
}

// ---------------------------------------------------------------------------
// Gather + RMSNorm(ln1): x, h_bf16, g, pos
// ---------------------------------------------------------------------------
__global__ __launch_bounds__(256) void gather_norm_kernel(
    const float* __restrict__ hidden, const float* __restrict__ gating,
    const int* __restrict__ sel, const int* __restrict__ posids,
    const float* __restrict__ lnw, float* __restrict__ xg,
    unsigned short* __restrict__ hb, float* __restrict__ gbuf,
    int* __restrict__ posbuf) {
  int i = blockIdx.x, tid = threadIdx.x;
  int s = sel[i];
  int b = s / Sc, t = s - b * Sc;
  const float* row = hidden + ((size_t)b * Sc + t) * Dc;
  float vals[4];
  float ss = 0.f;
#pragma unroll
  for (int c = 0; c < 4; c++) {
    float v = row[tid + 256 * c];
    vals[c] = v;
    ss += v * v;
  }
  __shared__ float red[256];
  red[tid] = ss;
  __syncthreads();
  for (int off = 128; off > 0; off >>= 1) {
    if (tid < off) red[tid] += red[tid + off];
    __syncthreads();
  }
  float rs = rsqrtf(red[0] / (float)Dc + 1e-6f);
#pragma unroll
  for (int c = 0; c < 4; c++) {
    int d = tid + 256 * c;
    xg[(size_t)i * Dc + d] = vals[c];
    hb[(size_t)i * Dc + d] = f2bf(vals[c] * rs * lnw[d]);
  }
  if (tid == 0) {
    gbuf[i] = gating[(size_t)b * Sc + t];
    posbuf[i] = posids[(size_t)b * Sc + t];
  }
}

// ---------------------------------------------------------------------------
// Plain RMSNorm(ln2): x1 f32 -> h2 bf16
// ---------------------------------------------------------------------------
__global__ __launch_bounds__(256) void rmsnorm_kernel(
    const float* __restrict__ X, const float* __restrict__ w,
    unsigned short* __restrict__ Ob) {
  int i = blockIdx.x, tid = threadIdx.x;
  const float* row = X + (size_t)i * Dc;
  float vals[4];
  float ss = 0.f;
#pragma unroll
  for (int c = 0; c < 4; c++) {
    float v = row[tid + 256 * c];
    vals[c] = v;
    ss += v * v;
  }
  __shared__ float red[256];
  red[tid] = ss;
  __syncthreads();
  for (int off = 128; off > 0; off >>= 1) {
    if (tid < off) red[tid] += red[tid + off];
    __syncthreads();
  }
  float rs = rsqrtf(red[0] / (float)Dc + 1e-6f);
#pragma unroll
  for (int c = 0; c < 4; c++) {
    int d = tid + 256 * c;
    Ob[(size_t)i * Dc + d] = f2bf(vals[c] * rs * w[d]);
  }
}

// ---------------------------------------------------------------------------
// WMMA GEMM with TDM-staged, double-buffered B tiles in LDS.
// Block = 128x64 output macro-tile (8 waves x 16 rows, shared tn).
// C[M,N] = A[M,Kd](bf16) * Bt[N,Kd](bf16)^T
// MODE 0: f32 out (+bias? +residual?)   MODE 1: bf16 out (+bias?)
// MODE 2: bf16 silu(out)                MODE 3: bf16 out * bf2f(other)
// ---------------------------------------------------------------------------
template <int MODE>
__global__ __launch_bounds__(256) void gemm_bf16_kernel(
    const unsigned short* __restrict__ A, const unsigned short* __restrict__ Bt,
    const float* __restrict__ bias, const float* __restrict__ residual,
    const unsigned short* __restrict__ other, float* __restrict__ Cf,
    unsigned short* __restrict__ Cb, int M, int N, int Kd) {
  __shared__ __align__(16) unsigned short ldsB[2][64 * 64];  // 2 x 8 KB
  const int lane = (int)(threadIdx.x & 31u);
  const int wave = (int)(threadIdx.x >> 5u);
  const int ntn = N >> 6;
  const int bm = blockIdx.x / ntn;          // 128-row block index
  const int tn = blockIdx.x - bm * ntn;     // 64-col block index
  const int tm = bm * 8 + wave;             // this wave's 16-row tile
  const int m = lane & 15, half = lane >> 4;

  const unsigned short* Arow = A + (size_t)(tm * 16 + m) * Kd;
  const unsigned short* Btile = Bt + (size_t)(tn * 64) * Kd;

  // Preload chunk 0 (TDM DMA -> LDS buffer 0)
  if (wave == 0) {
    tdm_load_b_tile(Btile, (unsigned)(size_t)&ldsB[0][0], Kd, N);
    __builtin_amdgcn_s_wait_tensorcnt(0);
  }
  __syncthreads();

  f32x8 acc[4] = {};
  const int nchunks = Kd >> 6;
  for (int c = 0; c < nchunks; c++) {
    const int kk = c << 6;
    // Kick off next chunk's DMA into the other buffer while we compute
    if (wave == 0 && c + 1 < nchunks) {
      tdm_load_b_tile(Btile + kk + 64,
                      (unsigned)(size_t)&ldsB[(c + 1) & 1][0], Kd, N);
    }
    if (kk + 64 < Kd) __builtin_prefetch(Arow + kk + 64, 0, 1);  // A prefetch
    const unsigned short* lb = &ldsB[c & 1][0];
    bf16x16 a0 = load_a_frag(Arow + kk, half);
    bf16x16 a1 = load_a_frag(Arow + kk + 32, half);
#pragma unroll
    for (int f = 0; f < 4; f++) {
      const unsigned short* rowp = lb + (size_t)(f * 16 + m) * 64;
      bf16x16 blo = load_b_frag(rowp, half);       // chunk k 0..31
      bf16x16 bhi = load_b_frag(rowp + 32, half);  // chunk k 32..63
      acc[f] = wmma_bf16(a0, blo, acc[f]);
      acc[f] = wmma_bf16(a1, bhi, acc[f]);
    }
    if (wave == 0) __builtin_amdgcn_s_wait_tensorcnt(0);
    __syncthreads();  // next buffer ready; all readers done with old buffer
  }

  // C/D layout: VGPR r, lanes 0-15 -> row r; lanes 16-31 -> row 8+r; col = lane&15
#pragma unroll
  for (int r = 0; r < 8; r++) {
    int row = tm * 16 + half * 8 + r;
#pragma unroll
    for (int f = 0; f < 4; f++) {
      int col = tn * 64 + f * 16 + m;
      size_t idx = (size_t)row * N + col;
      float v = acc[f][r];
      if (bias) v += bias[col];
      if constexpr (MODE == 0) {
        if (residual) v += residual[idx];
        Cf[idx] = v;
      } else if constexpr (MODE == 2) {
        Cb[idx] = f2bf(v / (1.f + __expf(-v)));   // silu
      } else if constexpr (MODE == 3) {
        Cb[idx] = f2bf(v * bf2f(other[idx]));
      } else {
        Cb[idx] = f2bf(v);
      }
    }
  }
}

// ---------------------------------------------------------------------------
// RoPE + pack: qf/kf/vf f32 [K][H*DH] -> qh/kh bf16 [H][K][DH], vth bf16 [H][DH][K]
// One thread per (token i, head h, pair j<32).
// ---------------------------------------------------------------------------
__global__ __launch_bounds__(256) void rope_pack_kernel(
    const float* __restrict__ qf, const float* __restrict__ kf,
    const float* __restrict__ vf, const int* __restrict__ pos,
    unsigned short* __restrict__ qh, unsigned short* __restrict__ kh,
    unsigned short* __restrict__ vth) {
  int idx = blockIdx.x * blockDim.x + threadIdx.x;
  if (idx >= Kc * Hc * 32) return;
  int j = idx & 31;
  int h = (idx >> 5) & (Hc - 1);
  int i = idx >> 9;
  float p = (float)pos[i];
  float inv = __powf(10000.0f, -(float)j / 32.0f);
  float ang = p * inv;
  float c = __cosf(ang), s = __sinf(ang);
  size_t base = (size_t)i * HDc + h * DHc;
  size_t qb = ((size_t)h * Kc + i) * DHc;
  float q1 = qf[base + j], q2 = qf[base + 32 + j];
  qh[qb + j] = f2bf(q1 * c - q2 * s);
  qh[qb + 32 + j] = f2bf(q2 * c + q1 * s);
  float k1 = kf[base + j], k2 = kf[base + 32 + j];
  kh[qb + j] = f2bf(k1 * c - k2 * s);
  kh[qb + 32 + j] = f2bf(k2 * c + k1 * s);
  float v1 = vf[base + j], v2 = vf[base + 32 + j];
  vth[((size_t)h * DHc + j) * Kc + i] = f2bf(v1);
  vth[((size_t)h * DHc + 32 + j) * Kc + i] = f2bf(v2);
}

// ---------------------------------------------------------------------------
// Flash attention: per-wave one (head, 16-query tile). Causal, online softmax.
// attnb bf16 [K][H*DH]
// ---------------------------------------------------------------------------
__global__ __launch_bounds__(256) void flash_attn_kernel(
    const unsigned short* __restrict__ qh, const unsigned short* __restrict__ kh,
    const unsigned short* __restrict__ vth, unsigned short* __restrict__ attnb) {
  __shared__ __align__(16) unsigned short plds[8][16 * 32];  // per-wave P tile
  const int lane = (int)(threadIdx.x & 31u);
  const int w = (int)(threadIdx.x >> 5u);
  const int task = blockIdx.x * 8 + w;
  const int h = task >> 7;       // / (Kc/16)
  const int qt = task & 127;
  const int q0 = qt * 16;
  const int m = lane & 15, half = lane >> 4;
  const int nl = m, hi = half;

  // Q fragments (16x64 split into two 16x32 A-frags)
  const unsigned short* qbase = qh + ((size_t)h * Kc + q0 + m) * DHc;
  BF16Frag aq0, aq1;
  aq0.u[0] = *(const u16x8*)(qbase + 16 * half);
  aq0.u[1] = *(const u16x8*)(qbase + 16 * half + 8);
  aq1.u[0] = *(const u16x8*)(qbase + 32 + 16 * half);
  aq1.u[1] = *(const u16x8*)(qbase + 32 + 16 * half + 8);

  f32x8 o0 = {}, o1 = {}, o2 = {}, o3 = {};
  float m8[8], l8[8];
#pragma unroll
  for (int r = 0; r < 8; r++) { m8[r] = -1e30f; l8[r] = 0.f; }
  unsigned short* pb = &plds[w][0];

  for (int kb = 0; kb < q0 + 16; kb += 32) {
    // K^T B-fragments: lane nl = key index; contiguous along d
    const unsigned short* kr0 = kh + ((size_t)h * Kc + kb + nl) * DHc + 16 * half;
    const unsigned short* kr1 = kr0 + (size_t)16 * DHc;  // keys +16
    BF16Frag b00, b01, b10, b11;
    b00.u[0] = *(const u16x8*)(kr0);      b00.u[1] = *(const u16x8*)(kr0 + 8);
    b01.u[0] = *(const u16x8*)(kr0 + 32); b01.u[1] = *(const u16x8*)(kr0 + 40);
    b10.u[0] = *(const u16x8*)(kr1);      b10.u[1] = *(const u16x8*)(kr1 + 8);
    b11.u[0] = *(const u16x8*)(kr1 + 32); b11.u[1] = *(const u16x8*)(kr1 + 40);
    f32x8 s0 = {}, s1 = {};
    s0 = wmma_bf16(aq0.v, b00.v, s0);
    s0 = wmma_bf16(aq1.v, b01.v, s0);
    s1 = wmma_bf16(aq0.v, b10.v, s1);
    s1 = wmma_bf16(aq1.v, b11.v, s1);

    // Online softmax over 32 keys (row = 16 lanes within half-wave)
#pragma unroll
    for (int r = 0; r < 8; r++) {
      int row = q0 + hi * 8 + r;
      int key0 = kb + nl, key1 = key0 + 16;
      float v0 = (key0 <= row) ? s0[r] * 0.125f : -1e30f;  // 1/sqrt(64)
      float v1 = (key1 <= row) ? s1[r] * 0.125f : -1e30f;
      float mx = fmaxf(v0, v1);
      mx = fmaxf(mx, __shfl_xor(mx, 1, 32));
      mx = fmaxf(mx, __shfl_xor(mx, 2, 32));
      mx = fmaxf(mx, __shfl_xor(mx, 4, 32));
      mx = fmaxf(mx, __shfl_xor(mx, 8, 32));
      float nm = fmaxf(m8[r], mx);
      float corr = __expf(m8[r] - nm);
      float p0 = __expf(v0 - nm), p1 = __expf(v1 - nm);
      float rs = p0 + p1;
      rs += __shfl_xor(rs, 1, 32);
      rs += __shfl_xor(rs, 2, 32);
      rs += __shfl_xor(rs, 4, 32);
      rs += __shfl_xor(rs, 8, 32);
      l8[r] = l8[r] * corr + rs;
      m8[r] = nm;
      o0[r] *= corr; o1[r] *= corr; o2[r] *= corr; o3[r] *= corr;
      pb[(hi * 8 + r) * 32 + nl] = f2bf(p0);
      pb[(hi * 8 + r) * 32 + 16 + nl] = f2bf(p1);
    }
    asm volatile("s_wait_dscnt 0" ::: "memory");

    // P (16x32) C-layout -> A-layout via LDS
    BF16Frag pa;
    const unsigned short* pr = pb + m * 32;
    pa.u[0] = *(const u16x8*)(pr + 16 * half);
    pa.u[1] = *(const u16x8*)(pr + 16 * half + 8);

    // V^T B-fragments: lane nl = output dim d (per 16-col chunk); contiguous keys
    const unsigned short* vb =
        vth + ((size_t)h * DHc + nl) * Kc + kb + 16 * half;
    BF16Frag bv;
    bv.u[0] = *(const u16x8*)(vb); bv.u[1] = *(const u16x8*)(vb + 8);
    o0 = wmma_bf16(pa.v, bv.v, o0);
    vb += (size_t)16 * Kc;
    bv.u[0] = *(const u16x8*)(vb); bv.u[1] = *(const u16x8*)(vb + 8);
    o1 = wmma_bf16(pa.v, bv.v, o1);
    vb += (size_t)16 * Kc;
    bv.u[0] = *(const u16x8*)(vb); bv.u[1] = *(const u16x8*)(vb + 8);
    o2 = wmma_bf16(pa.v, bv.v, o2);
    vb += (size_t)16 * Kc;
    bv.u[0] = *(const u16x8*)(vb); bv.u[1] = *(const u16x8*)(vb + 8);
    o3 = wmma_bf16(pa.v, bv.v, o3);
  }

#pragma unroll
  for (int r = 0; r < 8; r++) {
    float inv = 1.0f / l8[r];
    int row = q0 + hi * 8 + r;
    unsigned short* orow = attnb + (size_t)row * HDc + h * DHc + nl;
    orow[0] = f2bf(o0[r] * inv);
    orow[16] = f2bf(o1[r] * inv);
    orow[32] = f2bf(o2[r] * inv);
    orow[48] = f2bf(o3[r] * inv);
  }
}

// ---------------------------------------------------------------------------
// Output assembly
// ---------------------------------------------------------------------------
__global__ __launch_bounds__(256) void copy_kernel(const float* __restrict__ src,
                                                   float* __restrict__ dst,
                                                   size_t n) {
  size_t i = (size_t)blockIdx.x * blockDim.x + threadIdx.x;
  if (i < n) dst[i] = src[i];
}

__global__ __launch_bounds__(256) void scatter_kernel(
    const float* __restrict__ xg, const float* __restrict__ y,
    const float* __restrict__ gbuf, const int* __restrict__ sel,
    float* __restrict__ out) {
  size_t idx = (size_t)blockIdx.x * blockDim.x + threadIdx.x;
  if (idx >= (size_t)Kc * Dc) return;
  int i = (int)(idx / Dc);
  int d = (int)(idx - (size_t)i * Dc);
  int s = sel[i];
  int b = s / Sc, t = s - b * Sc;
  float x = xg[idx], yy = y[idx], g = gbuf[i];
  out[((size_t)b * Sc + t) * Dc + d] = x + (yy - x) * g;
}

// ---------------------------------------------------------------------------
extern "C" void kernel_launch(void* const* d_in, const int* in_sizes, int n_in,
                              void* d_out, int out_size, void* d_ws,
                              size_t ws_size, hipStream_t stream) {
  const float* hidden = (const float*)d_in[0];
  const float* gating = (const float*)d_in[1];
  const int* sel = (const int*)d_in[2];
  const int* posids = (const int*)d_in[3];
  const float* ln1w = (const float*)d_in[4];
  const float* ln2w = (const float*)d_in[5];
  const float* wq = (const float*)d_in[6];
  const float* bq = (const float*)d_in[7];
  const float* wk = (const float*)d_in[8];
  const float* bk = (const float*)d_in[9];
  const float* wv = (const float*)d_in[10];
  const float* bv = (const float*)d_in[11];
  const float* wo = (const float*)d_in[12];
  const float* wg = (const float*)d_in[13];
  const float* wu = (const float*)d_in[14];
  const float* wd = (const float*)d_in[15];
  float* out = (float*)d_out;

  char* wsp = (char*)d_ws;
  auto alloc = [&](size_t bytes) -> char* {
    char* p = wsp;
    wsp += (bytes + 255) & ~size_t(255);
    return p;
  };
  unsigned short* wqt = (unsigned short*)alloc((size_t)Dc * HDc * 2);
  unsigned short* wkt = (unsigned short*)alloc((size_t)Dc * HDc * 2);
  unsigned short* wvt = (unsigned short*)alloc((size_t)Dc * HDc * 2);
  unsigned short* wot = (unsigned short*)alloc((size_t)HDc * Dc * 2);
  unsigned short* wgt = (unsigned short*)alloc((size_t)Dc * Fc * 2);
  unsigned short* wut = (unsigned short*)alloc((size_t)Dc * Fc * 2);
  unsigned short* wdt = (unsigned short*)alloc((size_t)Fc * Dc * 2);
  float* xg = (float*)alloc((size_t)Kc * Dc * 4);
  unsigned short* hbf = (unsigned short*)alloc((size_t)Kc * Dc * 2);
  float* gbuf = (float*)alloc((size_t)Kc * 4);
  int* posbuf = (int*)alloc((size_t)Kc * 4);
  float* qf = (float*)alloc((size_t)Kc * HDc * 4);
  float* kf = (float*)alloc((size_t)Kc * HDc * 4);
  float* vf = (float*)alloc((size_t)Kc * HDc * 4);
  unsigned short* qhb = (unsigned short*)alloc((size_t)Kc * HDc * 2);
  unsigned short* khb = (unsigned short*)alloc((size_t)Kc * HDc * 2);
  unsigned short* vthb = (unsigned short*)alloc((size_t)Kc * HDc * 2);
  unsigned short* attnb = (unsigned short*)alloc((size_t)Kc * HDc * 2);
  float* x1 = (float*)alloc((size_t)Kc * Dc * 4);
  unsigned short* h2 = (unsigned short*)alloc((size_t)Kc * Dc * 2);
  unsigned short* gateb = (unsigned short*)alloc((size_t)Kc * Fc * 2);
  unsigned short* mlpb = (unsigned short*)alloc((size_t)Kc * Fc * 2);
  float* y = (float*)alloc((size_t)Kc * Dc * 4);

  // 1) copy hidden_states into output
  {
    size_t n = (size_t)Bc * Sc * Dc;
    copy_kernel<<<(unsigned)((n + 255) / 256), 256, 0, stream>>>(hidden, out, n);
  }
  // 2) convert + transpose weights to bf16
  convert_wt_kernel<<<(Dc * HDc) / 256, 256, 0, stream>>>(wq, wqt, Dc, HDc);
  convert_wt_kernel<<<(Dc * HDc) / 256, 256, 0, stream>>>(wk, wkt, Dc, HDc);
  convert_wt_kernel<<<(Dc * HDc) / 256, 256, 0, stream>>>(wv, wvt, Dc, HDc);
  convert_wt_kernel<<<(HDc * Dc) / 256, 256, 0, stream>>>(wo, wot, HDc, Dc);
  convert_wt_kernel<<<(Dc * Fc) / 256, 256, 0, stream>>>(wg, wgt, Dc, Fc);
  convert_wt_kernel<<<(Dc * Fc) / 256, 256, 0, stream>>>(wu, wut, Dc, Fc);
  convert_wt_kernel<<<(Fc * Dc) / 256, 256, 0, stream>>>(wd, wdt, Fc, Dc);
  // 3) gather + rmsnorm(ln1)
  gather_norm_kernel<<<Kc, 256, 0, stream>>>(hidden, gating, sel, posids, ln1w,
                                             xg, hbf, gbuf, posbuf);
  // 4) QKV projections (WMMA + TDM): blocks = (M/128)*(N/64) = 16*16 = 256
  gemm_bf16_kernel<0><<<256, 256, 0, stream>>>(hbf, wqt, bq, nullptr, nullptr,
                                               qf, nullptr, Kc, HDc, Dc);
  gemm_bf16_kernel<0><<<256, 256, 0, stream>>>(hbf, wkt, bk, nullptr, nullptr,
                                               kf, nullptr, Kc, HDc, Dc);
  gemm_bf16_kernel<0><<<256, 256, 0, stream>>>(hbf, wvt, bv, nullptr, nullptr,
                                               vf, nullptr, Kc, HDc, Dc);
  // 5) RoPE + pack to attention layouts
  rope_pack_kernel<<<(Kc * Hc * 32) / 256, 256, 0, stream>>>(qf, kf, vf, posbuf,
                                                             qhb, khb, vthb);
  // 6) flash attention: H*(K/16)=2048 wave-tasks -> 256 blocks of 8 waves
  flash_attn_kernel<<<(Hc * (Kc / 16)) / 8, 256, 0, stream>>>(qhb, khb, vthb,
                                                              attnb);
  // 7) O projection + residual x -> x1
  gemm_bf16_kernel<0><<<256, 256, 0, stream>>>(attnb, wot, nullptr, xg, nullptr,
                                               x1, nullptr, Kc, Dc, HDc);
  // 8) rmsnorm(ln2)
  rmsnorm_kernel<<<Kc, 256, 0, stream>>>(x1, ln2w, h2);
  // 9) MLP: gate (silu), up (*gate), down (+x1). gate/up blocks = 16*64 = 1024
  gemm_bf16_kernel<2><<<1024, 256, 0, stream>>>(h2, wgt, nullptr, nullptr,
                                                nullptr, nullptr, gateb, Kc, Fc,
                                                Dc);
  gemm_bf16_kernel<3><<<1024, 256, 0, stream>>>(h2, wut, nullptr, nullptr,
                                                gateb, nullptr, mlpb, Kc, Fc,
                                                Dc);
  gemm_bf16_kernel<0><<<256, 256, 0, stream>>>(mlpb, wdt, nullptr, x1, nullptr,
                                               y, nullptr, Kc, Dc, Fc);
  // 10) gated scatter back into output
  scatter_kernel<<<(Kc * Dc) / 256, 256, 0, stream>>>(xg, y, gbuf, sel, out);
  (void)in_sizes; (void)n_in; (void)out_size; (void)ws_size;
}
